// LSTM_decoder_6794638262640
// MI455X (gfx1250) — matrix operand order repaired and checked by reading, not compiled, not run on hardware
//
#include <hip/hip_runtime.h>
#include <hip/hip_bf16.h>

// ---------------------------------------------------------------------------
// LSTM decoder for MI455X (gfx1250, wave32, WMMA).
//   BATCH=1024, HIDDEN=512, LATENT=256, SEQ=256
// 64 persistent workgroups (16 batch rows each, WMMA M=16) run the whole
// 256-step recurrence: LDS-resident double-buffered h_t, register-resident
// c_t and x_proj, L2-resident pre-swizzled bf16 W_hh fragments streamed each
// step (LICM blocked via an asm barrier on a 0-offset so loads stay GLOBAL
// and in-loop). One s_barrier per step. v_wmma_f32_16x16x32_bf16 throughout.
// ---------------------------------------------------------------------------

typedef __bf16 bf16;
typedef __attribute__((ext_vector_type(16))) __bf16 v16bf;
typedef __attribute__((ext_vector_type(8)))  __bf16 v8bf;
typedef __attribute__((ext_vector_type(8)))  float  v8f;

#define HIDDEN 512
#define LATENT 256
#define SEQ    256

union FragBF { v16bf v; v8bf h[2]; };

__device__ __forceinline__ float fast_sig(float x) {
    return __builtin_amdgcn_rcpf(1.0f + __expf(-x));
}
__device__ __forceinline__ float fast_tanh(float x) {
    return 2.0f * __builtin_amdgcn_rcpf(1.0f + __expf(-2.0f * x)) - 1.0f;
}
__device__ __forceinline__ v8f wmma_bf16(const FragBF& a, const FragBF& b, v8f c) {
    return __builtin_amdgcn_wmma_f32_16x16x32_bf16(
        false, a.v, false, b.v, (short)0, c, false, false);
}

// ---------------------------------------------------------------------------
// Prep 1: hcb[b][0:512] = h[b], hcb[b][512:1024] = c[b]   (f32 -> bf16)
// ---------------------------------------------------------------------------
__global__ void concat_bf16_kernel(const float* __restrict__ h,
                                   const float* __restrict__ c,
                                   bf16* __restrict__ hcb) {
    int idx = blockIdx.x * 256 + threadIdx.x;   // 1024*1024 threads
    int b = idx >> 10;
    int k = idx & 1023;
    float v = (k < HIDDEN) ? h[b * HIDDEN + k] : c[b * HIDDEN + (k - HIDDEN)];
    hcb[idx] = (bf16)v;
}

// ---------------------------------------------------------------------------
// Prep 2: swizzle W [N][K] (f32 row-major, out = x @ W.T) into per-lane WMMA
// B-fragment order (bf16). Fragment (kc, nt): lane l holds column
// n = nt*16 + (l&15), K = kc*32 + (l>>4)*8 + (e&7) + (e>>3)*16, e = 0..15 (32B/lane).
// Flat layout: dst[(kc*NT + nt)*512 + l*16 + e].
// ---------------------------------------------------------------------------
__global__ void swizzle_b_kernel(const float* __restrict__ W,
                                 bf16* __restrict__ dst, int N, int K) {
    int idx  = blockIdx.x * 256 + threadIdx.x;  // N*K threads
    int e    = idx & 15;
    int l    = (idx >> 4) & 31;
    int frag = idx >> 9;
    int NT   = N >> 4;
    int kc   = frag / NT;
    int nt   = frag - kc * NT;
    int n    = nt * 16 + (l & 15);
    int k    = kc * 32 + (l >> 4) * 8 + (e & 7) + (e >> 3) * 16;
    dst[idx] = (bf16)W[n * K + k];
}

// ---------------------------------------------------------------------------
// Init GEMM: OUT[16x16 tile] = act(A @ B + bias).  One wave per output tile.
// Templated so the K loop fully unrolls into immediate-offset loads.
// MODE 0: +bias0+bias1, f32 out (x_proj)
// MODE 1: ELU(+bias0), bf16 out (h0)
// MODE 2: ELU(+bias0), f32 out (c0)
// ---------------------------------------------------------------------------
template <int NT, int KC, int MODE>
__global__ __launch_bounds__(256) void gemm_init_kernel(
    const bf16*  __restrict__ A, int lda,
    const bf16*  __restrict__ Bfr,
    const float* __restrict__ bias0,
    const float* __restrict__ bias1,
    float* __restrict__ outf, bf16* __restrict__ outb, int ldo) {

    int wave   = threadIdx.x >> 5;
    int l      = threadIdx.x & 31;
    int tile   = blockIdx.x * 8 + wave;
    int mt     = tile / NT;
    int nt     = tile - mt * NT;
    int lane16 = l & 15;
    int hi     = (l >> 4) * 8;     // C-layout row offset / A K offset

    const char* ab = (const char*)(A + (size_t)(mt * 16 + lane16) * lda + hi);
    const char* bb = (const char*)Bfr + (size_t)nt * 1024 + l * 32;

    v8f acc = {0.f, 0.f, 0.f, 0.f, 0.f, 0.f, 0.f, 0.f};
#pragma unroll
    for (int kc = 0; kc < KC; ++kc) {
        FragBF a, b;
        const v8bf* pa = (const v8bf*)(ab + kc * 64);
        a.h[0] = pa[0];            // K = hi + 0..7
        a.h[1] = pa[2];            // K = hi + 16..23
        const v8bf* pb = (const v8bf*)(bb + kc * (NT * 1024));
        b.h[0] = pb[0];
        b.h[1] = pb[1];
        acc = wmma_bf16(a, b, acc);
    }

    int ncol = nt * 16 + lane16;
    float bv = bias0[ncol];
    if (MODE == 0) bv += bias1[ncol];

#pragma unroll
    for (int r = 0; r < 8; ++r) {
        float v = acc[r] + bv;
        if (MODE != 0) v = (v > 0.f) ? v : (__expf(v) - 1.0f);   // ELU
        int row = mt * 16 + r + hi;
        if (MODE == 1) outb[(size_t)row * ldo + ncol] = (bf16)v;
        else           outf[(size_t)row * ldo + ncol] = v;
    }
}

// ---------------------------------------------------------------------------
// Recurrent kernel: 64 blocks x 512 threads (16 waves).
// Block owns batch rows [16*bid, 16*bid+16). Wave w owns gate N-tiles
// {w, w+16, w+32, w+48} = latent slice [16w,16w+16) of i/f/g/o, so the
// LSTM nonlinearity and c_t state are wave-local.
// ---------------------------------------------------------------------------
__global__ __launch_bounds__(512) void lstm_recurrent_kernel(
    const bf16*  __restrict__ Bwhh,   // [8 kc][64 nt][512] bf16 fragments
    const float* __restrict__ xproj,  // [1024][1024] f32
    const bf16*  __restrict__ h0,     // [1024][256] bf16 row-major
    const float* __restrict__ c0,     // [1024][256] f32
    float*       __restrict__ out) {  // [1024][256][256] f32

    __shared__ __align__(16) bf16 hbuf[2][16 * LATENT];   // 2 x 8KB

    const int b0     = blockIdx.x * 16;
    const int tid    = threadIdx.x;
    const int w      = tid >> 5;
    const int l      = tid & 31;
    const int lane16 = l & 15;
    const int hi     = (l >> 4) * 8;          // row offset in C layout
    const int lcol   = w * 16 + lane16;       // latent index owned by lane

    // stage h0 tile into LDS buffer 0
    for (int i = tid; i < 16 * LATENT; i += 512)
        hbuf[0][i] = h0[(size_t)(b0 + (i >> 8)) * LATENT + (i & 255)];

    // x_proj tiles (stay in registers for all 256 steps; also serve as the
    // WMMA C operand for the first K chunk each step -> no acc-init moves)
    v8f xp[4];
#pragma unroll
    for (int g = 0; g < 4; ++g) {
        int ncol = (w + 16 * g) * 16 + lane16;
#pragma unroll
        for (int r = 0; r < 8; ++r)
            xp[g][r] = xproj[(size_t)(b0 + r + hi) * 1024 + ncol];
    }
    // c_t tile in registers (C layout)
    v8f creg;
#pragma unroll
    for (int r = 0; r < 8; ++r)
        creg[r] = c0[(size_t)(b0 + r + hi) * LATENT + lcol];

    // Loop-invariant bases; all loads/stores below use immediate offsets.
    const char* bb  = (const char*)Bwhh + (w << 10) + (l << 5);         // B frags
    const char* ab0 = (const char*)&hbuf[0][0] + lane16 * 512 + hi * 2; // A reads
    const char* ab1 = (const char*)&hbuf[1][0] + lane16 * 512 + hi * 2;
    bf16* hn0 = &hbuf[0][hi * LATENT + lcol];                           // h writes
    bf16* hn1 = &hbuf[1][hi * LATENT + lcol];
    float* outp = out + (size_t)(b0 + hi) * (SEQ * LATENT) + lcol;      // out, t=0

    __syncthreads();

    for (int t = 0; t < SEQ; ++t) {
        const char* ab = (t & 1) ? ab1 : ab0;
        bf16*       hn = (t & 1) ? hn0 : hn1;

        // Optimization barrier on a zero OFFSET (not the pointer!): defeats
        // LICM of the loop-invariant W_hh loads (round 2 hoisted 256 VGPRs of
        // B and spilled), while keeping the base pointer's global address
        // space so loads lower to global_load_b128, not flat_load_b128
        // (flat also ties up DScnt and adds aperture checks).
        unsigned zoff = 0;
        asm volatile("" : "+v"(zoff));
        const char* bbm = bb + zoff;

        FragBF a, an, bc0, bc1, bc2, bc3, bn0, bn1, bn2, bn3;
        // prologue: fragments for K chunk 0
        { const v8bf* p = (const v8bf*)(ab);          a.h[0] = p[0]; a.h[1] = p[2]; }
        { const v8bf* p = (const v8bf*)(bbm);         bc0.h[0] = p[0]; bc0.h[1] = p[1]; }
        { const v8bf* p = (const v8bf*)(bbm + 16384); bc1.h[0] = p[0]; bc1.h[1] = p[1]; }
        { const v8bf* p = (const v8bf*)(bbm + 32768); bc2.h[0] = p[0]; bc2.h[1] = p[1]; }
        { const v8bf* p = (const v8bf*)(bbm + 49152); bc3.h[0] = p[0]; bc3.h[1] = p[1]; }

        v8f acc0, acc1, acc2, acc3;
#pragma unroll
        for (int kc = 0; kc < 8; ++kc) {
            // software pipeline: issue chunk kc+1 loads before chunk kc WMMAs
            if (kc < 7) {
                const char* kb = bbm + (kc + 1) * 65536;   // 64 frags * 1KB
                { const v8bf* p = (const v8bf*)(ab + (kc + 1) * 64); an.h[0] = p[0]; an.h[1] = p[2]; }
                { const v8bf* p = (const v8bf*)(kb);          bn0.h[0] = p[0]; bn0.h[1] = p[1]; }
                { const v8bf* p = (const v8bf*)(kb + 16384);  bn1.h[0] = p[0]; bn1.h[1] = p[1]; }
                { const v8bf* p = (const v8bf*)(kb + 32768);  bn2.h[0] = p[0]; bn2.h[1] = p[1]; }
                { const v8bf* p = (const v8bf*)(kb + 49152);  bn3.h[0] = p[0]; bn3.h[1] = p[1]; }
            }
            if (kc == 0) {
                acc0 = wmma_bf16(a, bc0, xp[0]);
                acc1 = wmma_bf16(a, bc1, xp[1]);
                acc2 = wmma_bf16(a, bc2, xp[2]);
                acc3 = wmma_bf16(a, bc3, xp[3]);
            } else {
                acc0 = wmma_bf16(a, bc0, acc0);
                acc1 = wmma_bf16(a, bc1, acc1);
                acc2 = wmma_bf16(a, bc2, acc2);
                acc3 = wmma_bf16(a, bc3, acc3);
            }
            a = an; bc0 = bn0; bc1 = bn1; bc2 = bn2; bc3 = bn3;  // SSA renames
        }

        // Wave-local LSTM epilogue; write h_t+1 (bf16->LDS) and output (f32).
#pragma unroll
        for (int r = 0; r < 8; ++r) {
            float ig = fast_sig(acc0[r]);
            float fg = fast_sig(acc1[r]);
            float gg = fast_tanh(acc2[r]);
            float og = fast_sig(acc3[r]);
            float cv = fg * creg[r] + ig * gg;
            creg[r] = cv;
            float hv = og * fast_tanh(cv);
            outp[r * (SEQ * LATENT)] = hv;     // r * 65536 floats (immediate)
            hn[r * LATENT] = (bf16)hv;
        }
        outp += LATENT;          // advance t
        __syncthreads();         // h_{t+1} complete; readers of h_t done
    }
}

// ---------------------------------------------------------------------------
// Launch
// ---------------------------------------------------------------------------
extern "C" void kernel_launch(void* const* d_in, const int* in_sizes, int n_in,
                              void* d_out, int out_size, void* d_ws, size_t ws_size,
                              hipStream_t stream) {
    const float* h   = (const float*)d_in[0];
    const float* c   = (const float*)d_in[1];
    // d_in[2] = seq_len (matches compile-time SEQ=256)
    const float* ihW = (const float*)d_in[3];
    const float* ihb = (const float*)d_in[4];
    const float* icW = (const float*)d_in[5];
    const float* icb = (const float*)d_in[6];
    const float* Wih = (const float*)d_in[7];
    const float* Whh = (const float*)d_in[8];
    const float* bih = (const float*)d_in[9];
    const float* bhh = (const float*)d_in[10];
    float* out = (float*)d_out;

    char* ws = (char*)d_ws;
    const size_t MB = 1024ull * 1024ull;
    bf16*  hcb   = (bf16*)(ws);                       // 2 MB  [1024][1024]
    bf16*  fWih  = (bf16*)(ws + 2 * MB);              // 1 MB  frags N=1024 K=512
    bf16*  fIhW  = (bf16*)(ws + 3 * MB);              // 512K  frags N=256 K=1024
    bf16*  fIcW  = (bf16*)(ws + 3 * MB + 512 * 1024); // 512K
    bf16*  fWhh  = (bf16*)(ws + 4 * MB);              // 512K  frags N=1024 K=256
    float* xproj = (float*)(ws + 4 * MB + 512 * 1024);// 4 MB  [1024][1024]
    bf16*  h0    = (bf16*)(ws + 8 * MB + 512 * 1024); // 512K  [1024][256]
    float* c0    = (float*)(ws + 9 * MB);             // 1 MB  [1024][256]
    // total 10 MB of workspace

    concat_bf16_kernel<<<4096, 256, 0, stream>>>(h, c, hcb);
    swizzle_b_kernel<<<2048, 256, 0, stream>>>(Wih, fWih, 4 * LATENT, HIDDEN);
    swizzle_b_kernel<<<1024, 256, 0, stream>>>(ihW, fIhW, LATENT, 2 * HIDDEN);
    swizzle_b_kernel<<<1024, 256, 0, stream>>>(icW, fIcW, LATENT, 2 * HIDDEN);
    swizzle_b_kernel<<<1024, 256, 0, stream>>>(Whh, fWhh, 4 * LATENT, LATENT);

    // x_proj = h @ W_ih.T + b_ih + b_hh  (M=1024,N=1024,K=512; A = first 512
    // cols of hcb, lda=1024)
    gemm_init_kernel<64, 16, 0><<<512, 256, 0, stream>>>(
        hcb, 1024, fWih, bih, bhh, xproj, nullptr, 1024);
    // h0 = ELU(hc @ init_h_W.T + b) -> bf16  (M=1024,N=256,K=1024)
    gemm_init_kernel<16, 32, 1><<<128, 256, 0, stream>>>(
        hcb, 1024, fIhW, ihb, nullptr, nullptr, h0, 256);
    // c0 = ELU(hc @ init_c_W.T + b) -> f32
    gemm_init_kernel<16, 32, 2><<<128, 256, 0, stream>>>(
        hcb, 1024, fIcW, icb, nullptr, c0, nullptr, 256);

    // persistent recurrence: 64 batch tiles x 16 waves, 256 steps
    lstm_recurrent_kernel<<<64, 512, 0, stream>>>(fWhh, xproj, h0, c0, out);
}
